// FlexibleGSPHAR_9981503996522
// MI455X (gfx1250) — matrix operand order repaired and checked by reading, not compiled, not run on hardware
//
#include <hip/hip_runtime.h>

#define BATCH 512
#define N 128
#define NSQ (N * N)
#define TWO_PI_Q 1.5707963267948966f  // 2*pi*0.25

typedef _Float16 v16h __attribute__((ext_vector_type(16)));
typedef float v8f __attribute__((ext_vector_type(8)));

union HFrag { v16h v; _Float16 e[16]; };
union FAcc  { v8f  v; float    e[8];  };

// ---------------------------------------------------------------- Aq = relu(A - A^T)
__global__ __launch_bounds__(256)
void aq_kernel(const float* __restrict__ A, float* __restrict__ Aq) {
    int idx = blockIdx.x * 256 + threadIdx.x;          // < 16384
    int n = idx >> 7, m = idx & 127;
    Aq[idx] = fmaxf(A[n * N + m] - A[m * N + n], 0.f);
}

// ---------------------------------------------------------------- Ad = 0.5*(|corr5|+|corr22|)*Aq
// One workgroup per batch. Centered lag data staged as f16 in LDS (K padded to 32),
// 64 16x16 output tiles computed with v_wmma_f32_16x16x32_f16 (8 tiles per wave, 8 waves).
__global__ __launch_bounds__(256)
void corr_ad_kernel(const float* __restrict__ x5g, const float* __restrict__ x22g,
                    const float* __restrict__ Aq, float* __restrict__ Ad) {
    __shared__ _Float16 Xc5[N][32];
    __shared__ _Float16 Xc22[N][32];
    __shared__ float sc5[N];
    __shared__ float sc22[N];

    const int b = blockIdx.x;
    const int tid = threadIdx.x;

    if (tid < N) {
        const int n = tid;
        {   // lag-5: center, pad, fold 1/((T-1)*std) into per-row scale
            float v[5]; float mean = 0.f;
            for (int t = 0; t < 5; ++t) { v[t] = x5g[((size_t)b * N + n) * 5 + t]; mean += v[t]; }
            mean *= 0.2f;
            float ss = 0.f;
            for (int t = 0; t < 5; ++t) { float d = v[t] - mean; ss += d * d; Xc5[n][t] = (_Float16)d; }
            for (int t = 5; t < 32; ++t) Xc5[n][t] = (_Float16)0.f;
            float stdv = sqrtf(ss * 0.25f);
            float inv = (stdv == 0.f) ? 1.f : (1.f / stdv);
            sc5[n] = inv * 0.5f;                 // inv / sqrt(T-1), T-1 = 4
        }
        {   // lag-22
            float v[22]; float mean = 0.f;
            for (int t = 0; t < 22; ++t) { v[t] = x22g[((size_t)b * N + n) * 22 + t]; mean += v[t]; }
            mean *= (1.f / 22.f);
            float ss = 0.f;
            for (int t = 0; t < 22; ++t) { float d = v[t] - mean; ss += d * d; Xc22[n][t] = (_Float16)d; }
            for (int t = 22; t < 32; ++t) Xc22[n][t] = (_Float16)0.f;
            float stdv = sqrtf(ss * (1.f / 21.f));
            float inv = (stdv == 0.f) ? 1.f : (1.f / stdv);
            sc22[n] = inv * rsqrtf(21.f);        // inv / sqrt(T-1), T-1 = 21
        }
    }
    __syncthreads();

    const int wave = tid >> 5;       // wave32: 8 waves per block
    const int lane = tid & 31;
    const int lh   = lane >> 4;      // lane half (0/1)
    const int l15  = lane & 15;

    for (int kt = 0; kt < 8; ++kt) {
        const int t  = wave * 8 + kt;    // tile 0..63
        const int tm = t >> 3;           // row tile
        const int tn = t & 7;            // col tile

        HFrag a5, b5, a22, b22;
        // A 16x32 f16 layout: lanes 0-15 M=0..15 K{0..7,16..23}; lanes 16-31 K{8..15,24..31}
        // B 32x16 f16 layout: lanes 0-15 K{0..7,16..23}; lanes 16-31 K{8..15,24..31}; N = lane&15
        #pragma unroll
        for (int j = 0; j < 16; ++j) {
            const int v = j >> 1, pos = j & 1;
            const int ka = ((v & 3) << 1) + pos + ((v >> 2) << 4) + (lh << 3);
            const int kb = ka;
            a5.e[j]  = Xc5 [tm * 16 + l15][ka];
            b5.e[j]  = Xc5 [tn * 16 + l15][kb];   // B = Xc^T  =>  B[k][n] = Xc[n][k]
            a22.e[j] = Xc22[tm * 16 + l15][ka];
            b22.e[j] = Xc22[tn * 16 + l15][kb];
        }

        FAcc c5, c22;
        c5.v  = (v8f){0.f, 0.f, 0.f, 0.f, 0.f, 0.f, 0.f, 0.f};
        c22.v = c5.v;
        c5.v  = __builtin_amdgcn_wmma_f32_16x16x32_f16(false, a5.v,  false, b5.v,
                                                       (short)0, c5.v,  false, false);
        c22.v = __builtin_amdgcn_wmma_f32_16x16x32_f16(false, a22.v, false, b22.v,
                                                       (short)0, c22.v, false, false);

        // C/D layout: VGPR v, lane -> (M = v + 8*(lane/16), N = lane&15)
        #pragma unroll
        for (int v = 0; v < 8; ++v) {
            const int row = tm * 16 + v + (lh << 3);
            const int col = tn * 16 + l15;
            const float cr5  = fabsf(c5.e[v])  * sc5[row]  * sc5[col];
            const float cr22 = fabsf(c22.e[v]) * sc22[row] * sc22[col];
            Ad[((size_t)b * N + row) * N + col] = 0.5f * (cr5 + cr22) * Aq[row * N + col];
        }
    }
}

// ---------------------------------------------------------------- softmax over axis 1 (columns)
__global__ __launch_bounds__(128)
void softmax_col_kernel(float* __restrict__ Ad) {
    const int b = blockIdx.x;
    const int m = threadIdx.x;
    const size_t base = (size_t)b * NSQ + m;
    float mx = -3.4e38f;
    for (int n = 0; n < N; ++n) mx = fmaxf(mx, Ad[base + (size_t)n * N]);
    float s = 0.f;
    for (int n = 0; n < N; ++n) {
        float e = expf(Ad[base + (size_t)n * N] - mx);
        Ad[base + (size_t)n * N] = e;
        s += e;
    }
    const float inv = 1.f / s;
    for (int n = 0; n < N; ++n) Ad[base + (size_t)n * N] *= inv;
}

// ---------------------------------------------------------------- dis = 1/sqrt(rowsum(A_s))
__global__ __launch_bounds__(128)
void degree_kernel(const float* __restrict__ Ad, float* __restrict__ dis) {
    const int b = blockIdx.x;
    const int n = threadIdx.x;
    const size_t rb = (size_t)b * NSQ + (size_t)n * N;
    const size_t cb = (size_t)b * NSQ + n;
    float rs = 0.f, cs = 0.f;
    for (int m = 0; m < N; ++m) { rs += Ad[rb + m]; cs += Ad[cb + (size_t)m * N]; }
    dis[b * N + n] = rsqrtf(0.5f * (rs + cs));
}

// ---------------------------------------------------------------- L (Hermitian magnetic Laplacian)
__global__ __launch_bounds__(256)
void build_l_kernel(const float* __restrict__ Ad, const float* __restrict__ dis,
                    float* __restrict__ Hre, float* __restrict__ Him) {
    const size_t idx = (size_t)blockIdx.x * 256 + threadIdx.x;  // < 512*16384
    const int b = (int)(idx / NSQ);
    const int rem = (int)(idx % NSQ);
    const int n = rem >> 7, m = rem & 127;
    const float ad  = Ad[(size_t)b * NSQ + n * N + m];
    const float adT = Ad[(size_t)b * NSQ + m * N + n];
    const float as_ = 0.5f * (ad + adT);
    const float th  = TWO_PI_Q * (ad - adT);
    const float mag = dis[b * N + n] * as_ * dis[b * N + m];
    float sn, cn; sincosf(th, &sn, &cn);
    Hre[idx] = ((n == m) ? 1.f : 0.f) - mag * cn;
    Him[idx] = -mag * sn;
}

// ---------------------------------------------------------------- batched complex Hermitian Jacobi
// One workgroup per batch matrix. H and V fully LDS-resident: 4 * 128*128 f32 = 256 KB
// (fits CDNA5's 320 KB/WGP LDS). Parallel round-robin ordering: 64 disjoint pairs per
// round, 127 rounds per sweep. Rotation J: J[pp]=c*e^{i phi}, J[pq]=s*e^{i phi},
// J[qp]=-s, J[qq]=c ;  H <- J^H H J ; V <- V J.
__global__ __launch_bounds__(256)
void jacobi_kernel(const float* __restrict__ Hre_g, const float* __restrict__ Him_g,
                   float* __restrict__ Vre_g, float* __restrict__ Vim_g, int sweeps) {
    extern __shared__ float sm[];
    float* Hr = sm;
    float* Hi = sm + NSQ;
    float* Vr = sm + 2 * NSQ;
    float* Vi = sm + 3 * NSQ;
    __shared__ int   sp[64], sq[64];
    __shared__ float scc[64], ssv[64], sur[64], sui[64];

    const int b = blockIdx.x;
    const int tid = threadIdx.x;
    const size_t base = (size_t)b * NSQ;

    for (int i = tid; i < NSQ; i += 256) {
        Hr[i] = Hre_g[base + i];
        Hi[i] = Him_g[base + i];
        int r = i >> 7, c = i & 127;
        Vr[i] = (r == c) ? 1.f : 0.f;
        Vi[i] = 0.f;
    }
    __syncthreads();

    for (int sweep = 0; sweep < sweeps; ++sweep) {
        for (int r = 0; r < 127; ++r) {
            // ---- compute 64 disjoint rotations (circle / round-robin pairing)
            if (tid < 64) {
                const int i = tid;
                int p, q;
                if (i == 0) { p = 127; q = r % 127; }
                else { p = (r + i) % 127; q = (r - i + 127) % 127; }
                if (p > q) { int t = p; p = q; q = t; }
                const float bre = Hr[p * N + q], bim = Hi[p * N + q];
                const float m2 = bre * bre + bim * bim;
                float c, s, ur, ui;
                if (m2 < 1e-24f) { c = 1.f; s = 0.f; ur = 1.f; ui = 0.f; }
                else {
                    const float mg = sqrtf(m2);
                    ur = bre / mg; ui = bim / mg;               // e^{i phi}
                    const float app = Hr[p * N + p], aqq = Hr[q * N + q];
                    const float tau = (aqq - app) / (2.f * mg);
                    const float tt = (tau >= 0.f ? 1.f : -1.f) /
                                     (fabsf(tau) + sqrtf(1.f + tau * tau));
                    c = rsqrtf(1.f + tt * tt);
                    s = tt * c;
                }
                sp[i] = p; sq[i] = q; scc[i] = c; ssv[i] = s; sur[i] = ur; sui[i] = ui;
            }
            __syncthreads();
            // ---- row update: H <- J^H H   (rows of disjoint pairs)
            for (int w = tid; w < 64 * N; w += 256) {
                const int i = w >> 7, j = w & 127;
                const int p = sp[i], q = sq[i];
                const float c = scc[i], s = ssv[i], ur = sur[i], ui = sui[i];
                const float pr = Hr[p * N + j], pi = Hi[p * N + j];
                const float qr = Hr[q * N + j], qi = Hi[q * N + j];
                const float epr = ur * pr + ui * pi;   // e^{-i phi} * row_p
                const float epi = ur * pi - ui * pr;
                Hr[p * N + j] = c * epr - s * qr;
                Hi[p * N + j] = c * epi - s * qi;
                Hr[q * N + j] = s * epr + c * qr;
                Hi[q * N + j] = s * epi + c * qi;
            }
            __syncthreads();
            // ---- column update: H <- H J ; V <- V J
            for (int w = tid; w < 64 * N; w += 256) {
                const int i = w >> 7, j = w & 127;
                const int p = sp[i], q = sq[i];
                const float c = scc[i], s = ssv[i], ur = sur[i], ui = sui[i];
                float pr = Hr[j * N + p], pi = Hi[j * N + p];
                float qr = Hr[j * N + q], qi = Hi[j * N + q];
                float epr = ur * pr - ui * pi;         // e^{+i phi} * col_p
                float epi = ur * pi + ui * pr;
                Hr[j * N + p] = c * epr - s * qr;
                Hi[j * N + p] = c * epi - s * qi;
                Hr[j * N + q] = s * epr + c * qr;
                Hi[j * N + q] = s * epi + c * qi;
                pr = Vr[j * N + p]; pi = Vi[j * N + p];
                qr = Vr[j * N + q]; qi = Vi[j * N + q];
                epr = ur * pr - ui * pi;
                epi = ur * pi + ui * pr;
                Vr[j * N + p] = c * epr - s * qr;
                Vi[j * N + p] = c * epi - s * qi;
                Vr[j * N + q] = s * epr + c * qr;
                Vi[j * N + q] = s * epi + c * qi;
            }
            __syncthreads();
        }
    }
    for (int i = tid; i < NSQ; i += 256) { Vre_g[base + i] = Vr[i]; Vim_g[base + i] = Vi[i]; }
}

// ---------------------------------------------------------------- spectral transform + MLP
// U_dega = V (eigenvector columns), U = V^H.
__global__ __launch_bounds__(128)
void spectral_kernel(const float* __restrict__ x1, const float* __restrict__ x5,
                     const float* __restrict__ x22,
                     const float* __restrict__ Vre, const float* __restrict__ Vim,
                     const float* __restrict__ w5, const float* __restrict__ w22,
                     const float* __restrict__ Wr, const float* __restrict__ br,
                     const float* __restrict__ Wi, const float* __restrict__ bi,
                     const float* __restrict__ W1, const float* __restrict__ b1,
                     const float* __restrict__ W2, const float* __restrict__ b2,
                     const float* __restrict__ W3, const float* __restrict__ b3,
                     float* __restrict__ out) {
    __shared__ float X[N][28];     // x1 | x5 | x22 concatenated over time
    __shared__ float ysr[N], ysi[N];

    const int b = blockIdx.x;
    const int n = threadIdx.x;

    X[n][0] = x1[(size_t)b * N + n];
    for (int t = 0; t < 5; ++t)  X[n][1 + t] = x5 [((size_t)b * N + n) * 5  + t];
    for (int t = 0; t < 22; ++t) X[n][6 + t] = x22[((size_t)b * N + n) * 22 + t];
    __syncthreads();

    // Z[n][t] = sum_m V[n][m] * X[m][t]   (complex * real)
    float zr[28], zi[28];
    #pragma unroll
    for (int t = 0; t < 28; ++t) { zr[t] = 0.f; zi[t] = 0.f; }
    const size_t rowb = (size_t)b * NSQ + (size_t)n * N;
    for (int m = 0; m < N; ++m) {
        const float vr = Vre[rowb + m], vi = Vim[rowb + m];
        #pragma unroll
        for (int t = 0; t < 28; ++t) { const float xv = X[m][t]; zr[t] += vr * xv; zi[t] += vi * xv; }
    }

    const float s1r = zr[0], s1i = zi[0];
    float c5r = 0.f, c5i = 0.f;
    for (int t = 0; t < 5; ++t)  { const float w = w5[n * 5 + t];   c5r += w * zr[1 + t]; c5i += w * zi[1 + t]; }
    float c22r = 0.f, c22i = 0.f;
    for (int t = 0; t < 22; ++t) { const float w = w22[n * 22 + t]; c22r += w * zr[6 + t]; c22i += w * zi[6 + t]; }

    ysr[n] = s1r * Wr[0] + c5r * Wr[1] + c22r * Wr[2] + br[0];
    ysi[n] = s1i * Wi[0] + c5i * Wi[1] + c22i * Wi[2] + bi[0];
    __syncthreads();

    // y[n] = sum_m conj(V[m][n]) * yspec[m]
    float ar = 0.f, ai = 0.f;
    const size_t colb = (size_t)b * NSQ + n;
    for (int m = 0; m < N; ++m) {
        const float vr = Vre[colb + (size_t)m * N], vi = Vim[colb + (size_t)m * N];
        ar += vr * ysr[m] + vi * ysi[m];
        ai += vr * ysi[m] - vi * ysr[m];
    }

    // MLP: [2] -> 16 -> 16 -> 1, relu everywhere
    float h1[16];
    #pragma unroll
    for (int k = 0; k < 16; ++k)
        h1[k] = fmaxf(ar * W1[k] + ai * W1[16 + k] + b1[k], 0.f);
    float h2[16];
    #pragma unroll
    for (int j = 0; j < 16; ++j) {
        float acc = b2[j];
        #pragma unroll
        for (int k = 0; k < 16; ++k) acc += h1[k] * W2[k * 16 + j];
        h2[j] = fmaxf(acc, 0.f);
    }
    float o = b3[0];
    #pragma unroll
    for (int j = 0; j < 16; ++j) o += h2[j] * W3[j];
    out[(size_t)b * N + n] = fmaxf(o, 0.f);
}

// ----------------------------------------------------------------
extern "C" void kernel_launch(void* const* d_in, const int* in_sizes, int n_in,
                              void* d_out, int out_size, void* d_ws, size_t ws_size,
                              hipStream_t stream) {
    (void)in_sizes; (void)n_in; (void)out_size; (void)ws_size;

    const float* x1  = (const float*)d_in[0];
    const float* x5  = (const float*)d_in[1];
    const float* x22 = (const float*)d_in[2];
    const float* A   = (const float*)d_in[3];
    const float* w5  = (const float*)d_in[4];
    const float* w22 = (const float*)d_in[5];
    const float* Wr  = (const float*)d_in[6];
    const float* br  = (const float*)d_in[7];
    const float* Wi  = (const float*)d_in[8];
    const float* bi  = (const float*)d_in[9];
    const float* W1  = (const float*)d_in[10];
    const float* b1  = (const float*)d_in[11];
    const float* W2  = (const float*)d_in[12];
    const float* b2  = (const float*)d_in[13];
    const float* W3  = (const float*)d_in[14];
    const float* b3  = (const float*)d_in[15];
    float* out = (float*)d_out;

    // workspace layout (floats); Vre aliases Ad (Ad is dead after build_l_kernel)
    float* ws  = (float*)d_ws;
    float* Aq  = ws;                                  // 16384
    float* Ad  = Aq  + NSQ;                           // 512*16384
    float* Hre = Ad  + (size_t)BATCH * NSQ;           // 512*16384
    float* Him = Hre + (size_t)BATCH * NSQ;           // 512*16384
    float* Vim = Him + (size_t)BATCH * NSQ;           // 512*16384
    float* dis = Vim + (size_t)BATCH * NSQ;           // 512*128
    float* Vre = Ad;                                  // alias

    aq_kernel<<<NSQ / 256, 256, 0, stream>>>(A, Aq);
    corr_ad_kernel<<<BATCH, 256, 0, stream>>>(x5, x22, Aq, Ad);
    softmax_col_kernel<<<BATCH, 128, 0, stream>>>(Ad);
    degree_kernel<<<BATCH, 128, 0, stream>>>(Ad, dis);
    build_l_kernel<<<(BATCH * NSQ) / 256, 256, 0, stream>>>(Ad, dis, Hre, Him);

    const size_t jacobi_lds = 4 * (size_t)NSQ * sizeof(float);   // 256 KB (CDNA5: 320 KB/WGP)
    jacobi_kernel<<<BATCH, 256, jacobi_lds, stream>>>(Hre, Him, Vre, Vim, 8);

    spectral_kernel<<<BATCH, 128, 0, stream>>>(x1, x5, x22, Vre, Vim, w5, w22,
                                               Wr, br, Wi, bi, W1, b1, W2, b2, W3, b3, out);
}
